// AdaptiveFeatureAlignment_12979391168734
// MI455X (gfx1250) — matrix operand order repaired and checked by reading, not compiled, not run on hardware
//
#include <hip/hip_runtime.h>
#include <hip/hip_bf16.h>
#include <stdint.h>

// ---------------------------------------------------------------------------
// AdaptiveFeatureAlignment for MI455X (gfx1250, wave32, WMMA)
//   B=8, C=256, H=W=80
//   conv1 (3x3, 256->64) + GN + SiLU      -> bf16 WMMA 16x16x32, K=2304
//   conv2 (3x3, 64->18)                   -> bf16 WMMA, K=576, N padded to 32
//   importance (1x1 256->32 silu, 32->9)  -> bf16 WMMA fused via LDS relayout
//   deformable 9-tap bilinear + einsum    -> coalesced NHWC f32 gathers
// ---------------------------------------------------------------------------

typedef __attribute__((ext_vector_type(16))) __bf16 v16bf;
typedef __attribute__((ext_vector_type(8)))  float  v8f;

#define NB  8
#define NC  256
#define NH  80
#define NW  80
#define NHW 6400
#define C4  64
#define C8  32
#define KK1 2304   // 9*256
#define KK2 576    // 9*64

union BF16x16 { v16bf v; uint4 q[2]; };

static __device__ __forceinline__ __bf16 f2bf(float f) {
  unsigned u = __builtin_bit_cast(unsigned, f);
  unsigned r = u + 0x7FFFu + ((u >> 16) & 1u);   // round-to-nearest-even
  unsigned short h = (unsigned short)(r >> 16);
  return __builtin_bit_cast(__bf16, h);
}

static __device__ __forceinline__ float sigm(float x) {
  return 1.0f / (1.0f + __expf(-x));
}

static __device__ __forceinline__ void fma4(float4& a, float s, float4 v) {
  a.x += s * v.x; a.y += s * v.y; a.z += s * v.z; a.w += s * v.w;
}

// branch-free padding: keep loaded dwords when valid, else zero (v_cndmask)
static __device__ __forceinline__ uint4 sel4(bool c, uint4 v) {
  uint4 z;
  z.x = c ? v.x : 0u; z.y = c ? v.y : 0u;
  z.z = c ? v.z : 0u; z.w = c ? v.w : 0u;
  return z;
}

// ---------------------------------------------------------------------------
// Pack x (NCHW f32) -> NHWC f32 (for sampling gathers) + NHWC bf16 (for WMMA)
// ---------------------------------------------------------------------------
__global__ __launch_bounds__(256) void pack_x_kernel(
    const float* __restrict__ x, float* __restrict__ xh, __bf16* __restrict__ xbf)
{
  int bp = blockIdx.x * blockDim.x + threadIdx.x;   // 51200
  if (bp >= NB * NHW) return;
  int b = bp / NHW;
  int p = bp - b * NHW;
  const float* src = x + (size_t)b * NC * NHW + p;
  float*  dh = xh  + (size_t)bp * NC;
  __bf16* db = xbf + (size_t)bp * NC;
  for (int c = 0; c < NC; ++c) {
    float v = src[(size_t)c * NHW];    // lanes: consecutive p -> coalesced
    dh[c] = v;
    db[c] = f2bf(v);
  }
}

// ---------------------------------------------------------------------------
// Pack weights into [N][K] bf16 (k = tap*Cin + ci), zero-pad N; zero GN stats
// ---------------------------------------------------------------------------
__global__ __launch_bounds__(256) void pack_w_kernel(
    const float* __restrict__ w1, const float* __restrict__ w2,
    const float* __restrict__ iw1, const float* __restrict__ iw2,
    __bf16* __restrict__ w1p, __bf16* __restrict__ w2p,
    __bf16* __restrict__ iw1p, __bf16* __restrict__ iw2p,
    float* __restrict__ stats)
{
  int tid  = blockIdx.x * blockDim.x + threadIdx.x;
  int nthr = gridDim.x * blockDim.x;
  if (tid < 128) stats[tid] = 0.0f;
  for (int i = tid; i < C4 * KK1; i += nthr) {        // w1: OIHW (64,256,3,3)
    int o = i / KK1, k = i - o * KK1;
    int t = k / NC, ci = k - t * NC;
    w1p[i] = f2bf(w1[((size_t)(o * NC + ci)) * 9 + t]);
  }
  for (int i = tid; i < 32 * KK2; i += nthr) {        // w2: (18,64,3,3), pad->32
    int o = i / KK2, k = i - o * KK2;
    int t = k / C4, ci = k - t * C4;
    w2p[i] = (o < 18) ? f2bf(w2[((size_t)(o * C4 + ci)) * 9 + t]) : f2bf(0.0f);
  }
  for (int i = tid; i < C8 * NC; i += nthr) {         // iw1: (32,256,1,1)
    iw1p[i] = f2bf(iw1[i]);
  }
  for (int i = tid; i < 16 * C8; i += nthr) {         // iw2: (9,32), pad->16
    int o = i >> 5, ci = i & 31;
    iw2p[i] = (o < 9) ? f2bf(iw2[o * C8 + ci]) : f2bf(0.0f);
  }
}

// ---------------------------------------------------------------------------
// conv1: one wave -> 16 pixels x 64 channels; 72 K-steps x 4 N-tiles of WMMA.
// All 10 fragment loads of a K-step are issued as one clause before the 4
// WMMAs so s_wait_loadcnt decrements overlap math with memory.
// Also accumulates GroupNorm sum/sumsq per (b, group) via shfl-tree + atomics.
// ---------------------------------------------------------------------------
__global__ __launch_bounds__(256) void conv1_kernel(
    const __bf16* __restrict__ xbf, const __bf16* __restrict__ w1p,
    const float* __restrict__ b1, float* __restrict__ hid_raw,
    float* __restrict__ stats)
{
  const int lane = threadIdx.x & 31;
  const int wv   = threadIdx.x >> 5;
  const int tile = blockIdx.x * 8 + wv;        // 0..3199
  const int r    = lane & 15;
  const int hi   = lane >> 4;

  const int bp0 = tile * 16;                   // 16 consecutive pixels, same row
  const int b   = bp0 / NHW;
  const int p0  = bp0 - b * NHW;
  const int y   = p0 / NW;
  const int xp  = (p0 - y * NW) + r;           // this lane's A-row pixel

  v8f acc[4];
  #pragma unroll
  for (int i = 0; i < 4; ++i)
    #pragma unroll
    for (int j = 0; j < 8; ++j) acc[i][j] = 0.0f;

  for (int t = 0; t < 9; ++t) {
    const int yy = y + (t / 3) - 1;
    const int xx = xp + (t % 3) - 1;
    const bool valid = (yy >= 0) && (yy < NH) && (xx >= 0) && (xx < NW);
    const int yyc = min(max(yy, 0), NH - 1);   // clamped addr, result masked
    const int xxc = min(max(xx, 0), NW - 1);
    const __bf16* arow = xbf + ((size_t)(b * NHW + yyc * NW + xxc)) * NC + hi * 8;
    for (int kc = 0; kc < NC; kc += 32) {
      // ---- issue every load of this K-step first (one clause) ----
      const __bf16* pa = arow + kc;
      uint4 aq0 = *(const uint4*)(pa);
      uint4 aq1 = *(const uint4*)(pa + 16);
      const int k = t * NC + kc;
      const __bf16* pb = w1p + (size_t)r * KK1 + k + hi * 8;
      BF16x16 bf[4];
      #pragma unroll
      for (int tn = 0; tn < 4; ++tn) {
        const __bf16* pbt = pb + (size_t)tn * 16 * KK1;
        bf[tn].q[0] = *(const uint4*)(pbt);
        bf[tn].q[1] = *(const uint4*)(pbt + 16);
      }
      BF16x16 a;
      a.q[0] = sel4(valid, aq0);
      a.q[1] = sel4(valid, aq1);
      // ---- then the 4 WMMAs ----
      #pragma unroll
      for (int tn = 0; tn < 4; ++tn)
        acc[tn] = __builtin_amdgcn_wmma_f32_16x16x32_bf16(
            false, a.v, false, bf[tn].v, (short)0, acc[tn], false, false);
    }
  }

  const int baserow = bp0 + hi * 8;
  #pragma unroll
  for (int tn = 0; tn < 4; ++tn) {
    const int c = tn * 16 + r;
    const float bias = b1[c];
    float s = 0.0f, sq = 0.0f;
    #pragma unroll
    for (int j = 0; j < 8; ++j) {
      float v = acc[tn][j] + bias;
      hid_raw[(size_t)(baserow + j) * C4 + c] = v;
      s += v; sq += v * v;
    }
    // reduce over lanes with same group (channels c0..c0+7): masks 1,2,4,16
    s  += __shfl_xor(s, 1);  s  += __shfl_xor(s, 2);  s  += __shfl_xor(s, 4);  s  += __shfl_xor(s, 16);
    sq += __shfl_xor(sq, 1); sq += __shfl_xor(sq, 2); sq += __shfl_xor(sq, 4); sq += __shfl_xor(sq, 16);
    if ((lane & 23) == 0) {                      // lanes 0 and 8
      const int g = tn * 2 + ((lane >> 3) & 1);
      atomicAdd(&stats[2 * (b * 8 + g)],     s);
      atomicAdd(&stats[2 * (b * 8 + g) + 1], sq);
    }
  }
}

// ---------------------------------------------------------------------------
// GN stats finalize: (sum, sumsq) -> (mean, rstd) over 8ch * 6400px
// ---------------------------------------------------------------------------
__global__ void gn_finalize_kernel(float* __restrict__ stats)
{
  int i = threadIdx.x;
  if (i < NB * 8) {
    const float inv = 1.0f / (8.0f * (float)NHW);
    float mean = stats[2 * i] * inv;
    float var  = stats[2 * i + 1] * inv - mean * mean;
    stats[2 * i]     = mean;
    stats[2 * i + 1] = rsqrtf(var + 1e-5f);
  }
}

// ---------------------------------------------------------------------------
// GN apply + SiLU -> bf16 NHWC hidden activations
// ---------------------------------------------------------------------------
__global__ __launch_bounds__(256) void gn_apply_kernel(
    const float* __restrict__ hid_raw, const float* __restrict__ stats,
    const float* __restrict__ gsc, const float* __restrict__ gbi,
    __bf16* __restrict__ hid_act)
{
  size_t i = (size_t)blockIdx.x * blockDim.x + threadIdx.x;  // 3,276,800
  if (i >= (size_t)NB * NHW * C4) return;
  int c = (int)(i & 63);
  size_t bp = i >> 6;
  int b = (int)(bp / NHW);
  int g = c >> 3;
  float mean = stats[2 * (b * 8 + g)];
  float rstd = stats[2 * (b * 8 + g) + 1];
  float v = hid_raw[i];
  float n = (v - mean) * rstd * gsc[c] + gbi[c];
  hid_act[i] = f2bf(n * sigm(n));
}

// ---------------------------------------------------------------------------
// conv2: 3x3, 64 -> 18 (N padded to 32); one wave -> 16 pixels
// ---------------------------------------------------------------------------
__global__ __launch_bounds__(256) void conv2_kernel(
    const __bf16* __restrict__ hid_act, const __bf16* __restrict__ w2p,
    const float* __restrict__ b2, float* __restrict__ offs)
{
  const int lane = threadIdx.x & 31;
  const int wv   = threadIdx.x >> 5;
  const int tile = blockIdx.x * 8 + wv;
  const int r    = lane & 15;
  const int hi   = lane >> 4;

  const int bp0 = tile * 16;
  const int b   = bp0 / NHW;
  const int p0  = bp0 - b * NHW;
  const int y   = p0 / NW;
  const int xp  = (p0 - y * NW) + r;

  v8f acc[2];
  #pragma unroll
  for (int i = 0; i < 2; ++i)
    #pragma unroll
    for (int j = 0; j < 8; ++j) acc[i][j] = 0.0f;

  for (int t = 0; t < 9; ++t) {
    const int yy = y + (t / 3) - 1;
    const int xx = xp + (t % 3) - 1;
    const bool valid = (yy >= 0) && (yy < NH) && (xx >= 0) && (xx < NW);
    const int yyc = min(max(yy, 0), NH - 1);
    const int xxc = min(max(xx, 0), NW - 1);
    const __bf16* arow = hid_act + ((size_t)(b * NHW + yyc * NW + xxc)) * C4 + hi * 8;
    #pragma unroll
    for (int kc = 0; kc < C4; kc += 32) {
      const __bf16* pa = arow + kc;
      uint4 aq0 = *(const uint4*)(pa);
      uint4 aq1 = *(const uint4*)(pa + 16);
      const int k = t * C4 + kc;
      const __bf16* pb = w2p + (size_t)r * KK2 + k + hi * 8;
      BF16x16 bf[2];
      #pragma unroll
      for (int tn = 0; tn < 2; ++tn) {
        const __bf16* pbt = pb + (size_t)tn * 16 * KK2;
        bf[tn].q[0] = *(const uint4*)(pbt);
        bf[tn].q[1] = *(const uint4*)(pbt + 16);
      }
      BF16x16 a;
      a.q[0] = sel4(valid, aq0);
      a.q[1] = sel4(valid, aq1);
      #pragma unroll
      for (int tn = 0; tn < 2; ++tn)
        acc[tn] = __builtin_amdgcn_wmma_f32_16x16x32_bf16(
            false, a.v, false, bf[tn].v, (short)0, acc[tn], false, false);
    }
  }

  const int baserow = bp0 + hi * 8;
  #pragma unroll
  for (int tn = 0; tn < 2; ++tn) {
    const int c = tn * 16 + r;
    if (c < 18) {
      const float bias = b2[c];
      #pragma unroll
      for (int j = 0; j < 8; ++j)
        offs[(size_t)(baserow + j) * 18 + c] = acc[tn][j] + bias;
    }
  }
}

// ---------------------------------------------------------------------------
// importance: silu(x @ iw1 + ib1) @ iw2 + ib2 -> sigmoid.  Layer-2 input is
// re-staged through LDS (WMMA D-layout -> A-layout) and fed to a second WMMA.
// ---------------------------------------------------------------------------
__global__ __launch_bounds__(256) void imp_kernel(
    const __bf16* __restrict__ xbf, const __bf16* __restrict__ iw1p,
    const float* __restrict__ ib1, const __bf16* __restrict__ iw2p,
    const float* __restrict__ ib2, float* __restrict__ impw)
{
  __shared__ __align__(16) __bf16 lds[8 * 16 * C8];   // 8 KB
  const int lane = threadIdx.x & 31;
  const int wv   = threadIdx.x >> 5;
  const int tile = blockIdx.x * 8 + wv;
  const int r    = lane & 15;
  const int hi   = lane >> 4;
  const int bp0  = tile * 16;

  v8f acc[2];
  #pragma unroll
  for (int i = 0; i < 2; ++i)
    #pragma unroll
    for (int j = 0; j < 8; ++j) acc[i][j] = 0.0f;

  const __bf16* arow = xbf + (size_t)(bp0 + r) * NC + hi * 8;
  #pragma unroll
  for (int kc = 0; kc < NC; kc += 32) {
    const __bf16* pa = arow + kc;
    BF16x16 a;
    a.q[0] = *(const uint4*)(pa);
    a.q[1] = *(const uint4*)(pa + 16);
    const __bf16* pb = iw1p + (size_t)r * NC + kc + hi * 8;
    BF16x16 bf[2];
    #pragma unroll
    for (int tn = 0; tn < 2; ++tn) {
      const __bf16* pbt = pb + (size_t)tn * 16 * NC;
      bf[tn].q[0] = *(const uint4*)(pbt);
      bf[tn].q[1] = *(const uint4*)(pbt + 16);
    }
    #pragma unroll
    for (int tn = 0; tn < 2; ++tn)
      acc[tn] = __builtin_amdgcn_wmma_f32_16x16x32_bf16(
          false, a.v, false, bf[tn].v, (short)0, acc[tn], false, false);
  }

  __bf16* l = lds + wv * 16 * C8;
  #pragma unroll
  for (int tn = 0; tn < 2; ++tn) {
    const int c = tn * 16 + r;
    const float bias = ib1[c];
    #pragma unroll
    for (int j = 0; j < 8; ++j) {
      float v = acc[tn][j] + bias;
      l[(hi * 8 + j) * C8 + c] = f2bf(v * sigm(v));
    }
  }
  __syncthreads();

  BF16x16 a2, b2f;
  const __bf16* pa2 = l + r * C8 + hi * 8;
  a2.q[0] = *(const uint4*)(pa2);
  a2.q[1] = *(const uint4*)(pa2 + 16);
  const __bf16* pb2 = iw2p + (size_t)r * C8 + hi * 8;
  b2f.q[0] = *(const uint4*)(pb2);
  b2f.q[1] = *(const uint4*)(pb2 + 16);
  v8f acc2;
  #pragma unroll
  for (int j = 0; j < 8; ++j) acc2[j] = 0.0f;
  acc2 = __builtin_amdgcn_wmma_f32_16x16x32_bf16(
      false, a2.v, false, b2f.v, (short)0, acc2, false, false);

  if (r < 9) {
    const float bias = ib2[r];
    #pragma unroll
    for (int j = 0; j < 8; ++j)
      impw[(size_t)(bp0 + hi * 8 + j) * 9 + r] = sigm(acc2[j] + bias);
  }
}

// ---------------------------------------------------------------------------
// Deformable sampling + einsum.  One wave = one pixel, lane l = channels
// [8l, 8l+8): each corner is one contiguous 1 KB read (two float4 loads per lane)
// from the NHWC f32 copy (L2-resident). Importance folded into corner weights.
// ---------------------------------------------------------------------------
__global__ __launch_bounds__(256) void sample_kernel(
    const float* __restrict__ xh, const float* __restrict__ offs,
    const float* __restrict__ impw, float* __restrict__ out)
{
  const int lane = threadIdx.x & 31;
  const int wv   = threadIdx.x >> 5;
  const int bp   = blockIdx.x * 8 + wv;       // 0..51199
  const int b    = bp / NHW;
  const int p    = bp - b * NHW;
  const int py0  = p / NW;
  const int px0  = p - py0 * NW;

  const float* ofs = offs + (size_t)bp * 18;
  const float* im  = impw + (size_t)bp * 9;
  const float* xb  = xh + (size_t)b * NHW * NC + lane * 8;

  float4 a0 = {0.f, 0.f, 0.f, 0.f};
  float4 a1 = {0.f, 0.f, 0.f, 0.f};

  #pragma unroll
  for (int k = 0; k < 9; ++k) {
    float px = (float)px0 + (float)(k / 3 - 1) + ofs[2 * k];
    float py = (float)py0 + (float)(k % 3 - 1) + ofs[2 * k + 1];
    px = fminf(fmaxf(px, 0.0f), (float)(NW - 1));
    py = fminf(fmaxf(py, 0.0f), (float)(NH - 1));
    float xf = floorf(px), yf = floorf(py);
    float wx = px - xf,    wy = py - yf;
    int xi = (int)xf, yi = (int)yf;
    int x1 = min(xi + 1, NW - 1);
    int y1 = min(yi + 1, NH - 1);
    float w = im[k];
    float w00 = w * (1.0f - wx) * (1.0f - wy);
    float w01 = w * wx * (1.0f - wy);
    float w10 = w * (1.0f - wx) * wy;
    float w11 = w * wx * wy;
    const float* p00 = xb + (size_t)(yi * NW + xi) * NC;
    const float* p01 = xb + (size_t)(yi * NW + x1) * NC;
    const float* p10 = xb + (size_t)(y1 * NW + xi) * NC;
    const float* p11 = xb + (size_t)(y1 * NW + x1) * NC;
    float4 v00a = *(const float4*)(p00);
    float4 v00b = *(const float4*)(p00 + 4);
    float4 v01a = *(const float4*)(p01);
    float4 v01b = *(const float4*)(p01 + 4);
    float4 v10a = *(const float4*)(p10);
    float4 v10b = *(const float4*)(p10 + 4);
    float4 v11a = *(const float4*)(p11);
    float4 v11b = *(const float4*)(p11 + 4);
    fma4(a0, w00, v00a); fma4(a1, w00, v00b);
    fma4(a0, w01, v01a); fma4(a1, w01, v01b);
    fma4(a0, w10, v10a); fma4(a1, w10, v10b);
    fma4(a0, w11, v11a); fma4(a1, w11, v11b);
  }

  float* ob = out + ((size_t)b * NC + lane * 8) * NHW + p;   // NCHW output
  ob[0 * NHW] = a0.x; ob[1 * NHW] = a0.y; ob[2 * NHW] = a0.z; ob[3 * NHW] = a0.w;
  ob[4 * NHW] = a1.x; ob[5 * NHW] = a1.y; ob[6 * NHW] = a1.z; ob[7 * NHW] = a1.w;
}

// ---------------------------------------------------------------------------
extern "C" void kernel_launch(void* const* d_in, const int* in_sizes, int n_in,
                              void* d_out, int out_size, void* d_ws, size_t ws_size,
                              hipStream_t stream)
{
  (void)in_sizes; (void)n_in; (void)out_size; (void)ws_size;

  const float* x   = (const float*)d_in[0];
  const float* w1  = (const float*)d_in[1];
  const float* b1  = (const float*)d_in[2];
  const float* gsc = (const float*)d_in[3];
  const float* gbi = (const float*)d_in[4];
  const float* w2  = (const float*)d_in[5];
  const float* b2  = (const float*)d_in[6];
  const float* iw1 = (const float*)d_in[7];
  const float* ib1 = (const float*)d_in[8];
  const float* iw2 = (const float*)d_in[9];
  const float* ib2 = (const float*)d_in[10];
  float* out = (float*)d_out;

  char* w = (char*)d_ws;
  float*  xh      = (float*) w; w += (size_t)NB * NHW * NC * 4;   // 52.4 MB
  __bf16* xbf     = (__bf16*)w; w += (size_t)NB * NHW * NC * 2;   // 26.2 MB
  float*  hid_raw = (float*) w; w += (size_t)NB * NHW * C4 * 4;   // 13.1 MB
  __bf16* hid_act = (__bf16*)w; w += (size_t)NB * NHW * C4 * 2;   //  6.6 MB
  float*  offsb   = (float*) w; w += (size_t)NB * NHW * 18 * 4;   //  3.7 MB
  float*  impw    = (float*) w; w += (size_t)NB * NHW * 9 * 4;    //  1.8 MB
  __bf16* w1p     = (__bf16*)w; w += (size_t)C4 * KK1 * 2;
  __bf16* w2p     = (__bf16*)w; w += (size_t)32 * KK2 * 2;
  __bf16* iw1p    = (__bf16*)w; w += (size_t)C8 * NC * 2;
  __bf16* iw2p    = (__bf16*)w; w += (size_t)16 * C8 * 2;
  float*  stats   = (float*) w; w += 512;

  pack_w_kernel<<<64, 256, 0, stream>>>(w1, w2, iw1, iw2, w1p, w2p, iw1p, iw2p, stats);
  pack_x_kernel<<<(NB * NHW) / 256, 256, 0, stream>>>(x, xh, xbf);
  conv1_kernel<<<(NB * NHW / 16) / 8, 256, 0, stream>>>(xbf, w1p, b1, hid_raw, stats);
  gn_finalize_kernel<<<1, 64, 0, stream>>>(stats);
  gn_apply_kernel<<<(int)(((size_t)NB * NHW * C4 + 255) / 256), 256, 0, stream>>>(
      hid_raw, stats, gsc, gbi, hid_act);
  conv2_kernel<<<(NB * NHW / 16) / 8, 256, 0, stream>>>(hid_act, w2p, b2, offsb);
  imp_kernel<<<(NB * NHW / 16) / 8, 256, 0, stream>>>(xbf, iw1p, ib1, iw2p, ib2, impw);
  sample_kernel<<<(NB * NHW) / 8, 256, 0, stream>>>(xh, offsb, impw, out);
}
